// Attention_962072674618
// MI455X (gfx1250) — compile-verified, matrix operand order checked by reading, not run on hardware
//
#include <hip/hip_runtime.h>
#include <hip/hip_bf16.h>

// ---------------------------------------------------------------------------
// Sliding-window gated attention for MI455X (gfx1250, wave32, WMMA).
// GEMMs: v_wmma_f32_16x16x32_bf16 with async global->LDS double buffering.
// ---------------------------------------------------------------------------

typedef __bf16 bf16;
typedef __attribute__((ext_vector_type(16))) __bf16 v16bf;
typedef __attribute__((ext_vector_type(8)))  __bf16 v8bf;
typedef __attribute__((ext_vector_type(8)))  float  v8f;

#define NB   2048
#define DIMB 1024
#define HEADSB 8
#define DHB  64
#define DIB  512
#define WIN  512

// ---------------------------------------------------------------------------
// Fragment loader: 16x32 bf16 tile (A-layout), rows of 32 contiguous bf16.
// Lane l holds row (l&15); element e maps to k = (e/8)*16 + (l/16)*8 + (e%8).
// Both 8-element halves are single 16B vector loads.
// ---------------------------------------------------------------------------
__device__ inline v16bf frag_ab(const bf16* p, int stride) {
  int lane = threadIdx.x & 31;
  int row  = lane & 15;
  int kh   = (lane >> 4) << 3;           // 0 or 8
  const bf16* q = p + row * stride + kh;
  v8bf lo = *(const v8bf*)(q);
  v8bf hi = *(const v8bf*)(q + 16);
  return __builtin_shufflevector(lo, hi, 0,1,2,3,4,5,6,7,8,9,10,11,12,13,14,15);
}

// ---------------------------------------------------------------------------
// Async global -> LDS 16B copy (GLOBAL_LOAD_ASYNC_TO_LDS_B128, ASYNCcnt).
// LDS operand is the wave-relative LDS byte address (low 32b of flat addr).
// ---------------------------------------------------------------------------
__device__ inline void async_ld16(bf16* lds_dst, const bf16* gsrc) {
  unsigned lds = (unsigned)(unsigned long long)(size_t)lds_dst;
  unsigned long long ga = (unsigned long long)(size_t)gsrc;
  asm volatile("global_load_async_to_lds_b128 %0, %1, off"
               :: "v"(lds), "v"(ga) : "memory");
}

// ---------------------------------------------------------------------------
// f32 -> bf16 copy
// ---------------------------------------------------------------------------
__global__ void k_convert(const float* __restrict__ src, bf16* __restrict__ dst, int n) {
  int i = blockIdx.x * blockDim.x + threadIdx.x;
  if (i < n) dst[i] = (bf16)src[i];
}

// f32 [K][N] -> bf16 [N][K] (pre-transposed weights so WMMA B frags are contiguous)
__global__ void k_transpose(const float* __restrict__ src, bf16* __restrict__ dst,
                            int K, int N) {
  int i = blockIdx.x * blockDim.x + threadIdx.x;
  if (i < K * N) {
    int k = i / N, n = i - k * N;
    dst[(size_t)n * K + k] = (bf16)src[i];
  }
}

// ---------------------------------------------------------------------------
// gates = sigmoid(tokens @ Wg), mix = sigmoid(tokens @ Wmix); stored [token][8]
// ---------------------------------------------------------------------------
__global__ void k_gates(const float* __restrict__ tokens,
                        const float* __restrict__ Wg,
                        const float* __restrict__ Wmix,
                        float* __restrict__ gates, float* __restrict__ mix) {
  int t    = blockIdx.x;                 // 0 .. B*N-1
  int tid  = threadIdx.x;                // 0 .. 127
  int o    = tid >> 3;                   // 0..15
  int part = tid & 7;
  const float* w = (o < 8) ? Wg : Wmix;
  int h = o & 7;
  const float* x = tokens + (size_t)t * DIMB;
  float sum = 0.f;
  for (int k = part; k < DIMB; k += 8) sum += x[k] * w[k * HEADSB + h];
  #pragma unroll
  for (int off = 1; off < 8; off <<= 1) sum += __shfl_xor(sum, off, 8);
  if (part == 0) {
    float v = 1.f / (1.f + __expf(-sum));
    if (o < 8) gates[(size_t)t * HEADSB + h] = v;
    else       mix  [(size_t)t * HEADSB + h] = v;
  }
}

// ---------------------------------------------------------------------------
// WMMA GEMM: C[M][N] = A[M][K](bf16 row-major) @ Bt[N][K](bf16)^T
// BM=128, BN=64, BK=32; 128 threads = 4 waves; wave owns 2 M-tiles x 4 N-tiles
// (8 WMMAs / K-step). Global->LDS via async-to-LDS copies, LDS double-buffered:
// issue tile s+1, s_wait_asynccnt 6 (6 in-order ops/thread/tile => tile s done),
// barrier, 8 WMMAs, barrier. M%128==0, N%64==0, K%32==0 for all uses here.
// ---------------------------------------------------------------------------
__global__ void k_gemm(const bf16* __restrict__ A, const bf16* __restrict__ Bt,
                       float* __restrict__ C, int M, int N, int K) {
  __shared__ __align__(16) bf16 As[2][128][40];
  __shared__ __align__(16) bf16 Bs[2][64][40];
  int tid  = threadIdx.x;
  int lane = tid & 31;
  int wave = tid >> 5;
  int m0 = blockIdx.y * 128;
  int n0 = blockIdx.x * 64;

  v8f acc[2][4] = {};

  auto prefetch = [&](int s) {
    int k0  = s * 32;
    int buf = s & 1;
    #pragma unroll
    for (int i = 0; i < 4; ++i) {                    // A: 128x32 = 512 chunks
      int idx = tid + i * 128;
      int r = idx >> 2, c8 = (idx & 3) << 3;
      async_ld16(&As[buf][r][c8], &A[(size_t)(m0 + r) * K + k0 + c8]);
    }
    #pragma unroll
    for (int i = 0; i < 2; ++i) {                    // B: 64x32 = 256 chunks
      int idx = tid + i * 128;
      int r = idx >> 2, c8 = (idx & 3) << 3;
      async_ld16(&Bs[buf][r][c8], &Bt[(size_t)(n0 + r) * K + k0 + c8]);
    }
  };

  int steps = K >> 5;
  prefetch(0);
  for (int s = 0; s < steps; ++s) {
    if (s + 1 < steps) {
      prefetch(s + 1);
      asm volatile("s_wait_asynccnt 0x6" ::: "memory");
    } else {
      asm volatile("s_wait_asynccnt 0x0" ::: "memory");
    }
    __syncthreads();
    int buf = s & 1;
    v16bf a0 = frag_ab(&As[buf][wave * 32][0],      40);
    v16bf a1 = frag_ab(&As[buf][wave * 32 + 16][0], 40);
    #pragma unroll
    for (int j = 0; j < 4; ++j) {
      v16bf b = frag_ab(&Bs[buf][j * 16][0], 40);
      acc[0][j] = __builtin_amdgcn_wmma_f32_16x16x32_bf16(false, a0, false, b,
                                                          (short)0, acc[0][j], false, false);
      acc[1][j] = __builtin_amdgcn_wmma_f32_16x16x32_bf16(false, a1, false, b,
                                                          (short)0, acc[1][j], false, false);
    }
    __syncthreads();
  }

  #pragma unroll
  for (int t = 0; t < 2; ++t) {
    int mbase = m0 + wave * 32 + t * 16;
    #pragma unroll
    for (int j = 0; j < 4; ++j)
      #pragma unroll
      for (int r = 0; r < 8; ++r) {
        int row = mbase + r + ((lane >> 4) << 3);
        int col = n0 + j * 16 + (lane & 15);
        C[(size_t)row * N + col] = acc[t][j][r];
      }
  }
}

// ---------------------------------------------------------------------------
// QKV epilogue: scale+RoPE on q, RoPE on k, value-residual lerp on v.
//   qs, ks : bf16 [B][H][N][64]     (rows contiguous in dh)
//   vT     : bf16 [B][H][64][N]     (rows contiguous in keys -> B-frag friendly)
// ---------------------------------------------------------------------------
__global__ void k_qkv_epi(const float* __restrict__ qraw,
                          const float* __restrict__ kvraw,
                          const float* __restrict__ vres,
                          const float* __restrict__ mix,
                          bf16* __restrict__ qs, bf16* __restrict__ ks,
                          bf16* __restrict__ vT) {
  int idx = blockIdx.x * blockDim.x + threadIdx.x;
  if (idx >= 2 * HEADSB * NB * 32) return;
  int p = idx & 31;
  int n = (idx >> 5) & (NB - 1);
  int h = (idx >> 16) & 7;
  int b = idx >> 19;

  float invf = __powf(10000.f, -(float)(2 * p) / 64.f);
  float th = (float)n * invf;
  float c = __cosf(th), s = __sinf(th);

  size_t tok = (size_t)(b * NB + n);
  size_t bh  = (size_t)(b * HEADSB + h);
  size_t bhn = bh * NB + n;
  int col = h * DHB + 2 * p;

  const float sc = 0.125f;               // DH^-0.5
  float q0 = qraw[tok * DIB + col] * sc;
  float q1 = qraw[tok * DIB + col + 1] * sc;
  qs[bhn * DHB + 2 * p]     = (bf16)(q0 * c - q1 * s);
  qs[bhn * DHB + 2 * p + 1] = (bf16)(q1 * c + q0 * s);

  float k0 = kvraw[tok * (2 * DIB) + col];
  float k1 = kvraw[tok * (2 * DIB) + col + 1];
  ks[bhn * DHB + 2 * p]     = (bf16)(k0 * c - k1 * s);
  ks[bhn * DHB + 2 * p + 1] = (bf16)(k1 * c + k0 * s);

  float v0 = kvraw[tok * (2 * DIB) + DIB + col];
  float v1 = kvraw[tok * (2 * DIB) + DIB + col + 1];
  float mx = mix[tok * HEADSB + h];
  const float* rp = vres + (bh * NB + n) * DHB;
  v0 = v0 + (rp[2 * p]     - v0) * mx;
  v1 = v1 + (rp[2 * p + 1] - v1) * mx;
  vT[(bh * DHB + 2 * p)     * NB + n] = (bf16)v0;
  vT[(bh * DHB + 2 * p + 1) * NB + n] = (bf16)v1;
}

// ---------------------------------------------------------------------------
// Flash attention, sliding window [i-512, i]. grid (N/64, B*H), 128 threads.
// Wave owns a 16-row Q tile; KV streamed in 32-key tiles.
//   S  : two 16x16 halves, 2 WMMAs each (dh split 32+32)
//   P  : exp'd scores staged through wave-private LDS (manual s_wait_dscnt —
//        cross-lane dependence inside one wave), reloaded as A fragment
//   O += P @ V : 4 WMMAs, f32 accumulators, streaming softmax rescale
// ---------------------------------------------------------------------------
__global__ void k_flash(const bf16* __restrict__ qs, const bf16* __restrict__ ks,
                        const bf16* __restrict__ vT, const float* __restrict__ gates,
                        bf16* __restrict__ outg) {
  int bh = blockIdx.y;
  int b = bh >> 3, h = bh & 7;
  int lane = threadIdx.x & 31;
  int wave = threadIdx.x >> 5;
  int qb = blockIdx.x * 64 + wave * 16;

  const bf16* qp = qs + (size_t)bh * NB * DHB;
  const bf16* kp = ks + (size_t)bh * NB * DHB;
  const bf16* vp = vT + (size_t)bh * DHB * NB;

  __shared__ __align__(16) bf16 Pl[4][16][32];

  v16bf qa0 = frag_ab(qp + (size_t)qb * DHB,      DHB);   // dh 0..31
  v16bf qa1 = frag_ab(qp + (size_t)qb * DHB + 32, DHB);   // dh 32..63

  v8f O[4] = {};
  float mrun[8], lrun[8];
  #pragma unroll
  for (int r = 0; r < 8; ++r) { mrun[r] = -1e30f; lrun[r] = 0.f; }

  int jstart = qb - WIN; if (jstart < 0) jstart = 0;
  jstart &= ~31;
  int jend = qb + 15;

  for (int j0 = jstart; j0 <= jend; j0 += 32) {
    v8f s[2];
    #pragma unroll
    for (int ht = 0; ht < 2; ++ht) {
      const bf16* kb = kp + (size_t)(j0 + ht * 16) * DHB;
      v16bf kb0 = frag_ab(kb,      DHB);
      v16bf kb1 = frag_ab(kb + 32, DHB);
      v8f sv = {};
      sv = __builtin_amdgcn_wmma_f32_16x16x32_bf16(false, qa0, false, kb0,
                                                   (short)0, sv, false, false);
      sv = __builtin_amdgcn_wmma_f32_16x16x32_bf16(false, qa1, false, kb1,
                                                   (short)0, sv, false, false);
      s[ht] = sv;
    }

    int colj = lane & 15;
    #pragma unroll
    for (int r = 0; r < 8; ++r) {
      int row = r + ((lane >> 4) << 3);
      int i = qb + row;
      float s0 = s[0][r]; { int d = i - (j0 + colj);      if (d < 0 || d > WIN) s0 = -1e30f; }
      float s1 = s[1][r]; { int d = i - (j0 + 16 + colj); if (d < 0 || d > WIN) s1 = -1e30f; }

      float tm = fmaxf(s0, s1);
      #pragma unroll
      for (int off = 1; off < 16; off <<= 1) tm = fmaxf(tm, __shfl_xor(tm, off, 16));
      float mnew = fmaxf(mrun[r], tm);
      float corr = __expf(mrun[r] - mnew);
      mrun[r] = mnew;

      float p0 = __expf(s0 - mnew), p1 = __expf(s1 - mnew);
      float rs = p0 + p1;
      #pragma unroll
      for (int off = 1; off < 16; off <<= 1) rs += __shfl_xor(rs, off, 16);
      lrun[r] = lrun[r] * corr + rs;
      #pragma unroll
      for (int j = 0; j < 4; ++j) O[j][r] *= corr;

      Pl[wave][row][colj]      = (bf16)p0;
      Pl[wave][row][16 + colj] = (bf16)p1;
    }

    // cross-lane LDS dependence inside one wave: wait on the DS counter
    asm volatile("s_wait_dscnt 0x0" ::: "memory");

    v16bf pa = frag_ab(&Pl[wave][0][0], 32);
    #pragma unroll
    for (int jj = 0; jj < 4; ++jj) {
      v16bf vb = frag_ab(vp + (size_t)(jj * 16) * NB + j0, NB);
      O[jj] = __builtin_amdgcn_wmma_f32_16x16x32_bf16(false, pa, false, vb,
                                                      (short)0, O[jj], false, false);
    }
  }

  #pragma unroll
  for (int r = 0; r < 8; ++r) {
    int m = qb + r + ((lane >> 4) << 3);
    float g = gates[(size_t)(b * NB + m) * HEADSB + h] / lrun[r];
    #pragma unroll
    for (int jj = 0; jj < 4; ++jj) {
      outg[(size_t)(b * NB + m) * DIB + h * DHB + jj * 16 + (lane & 15)] =
          (bf16)(O[jj][r] * g);
    }
  }
}

// ---------------------------------------------------------------------------
// Host launch
// ---------------------------------------------------------------------------
static inline size_t al256(size_t x) { return (x + 255) & ~(size_t)255; }

extern "C" void kernel_launch(void* const* d_in, const int* in_sizes, int n_in,
                              void* d_out, int out_size, void* d_ws, size_t ws_size,
                              hipStream_t stream) {
  const float* tokens = (const float*)d_in[0];
  const float* vres   = (const float*)d_in[1];
  const float* Wq     = (const float*)d_in[2];
  const float* Wkv    = (const float*)d_in[3];
  const float* Wo     = (const float*)d_in[4];
  const float* Wg     = (const float*)d_in[5];
  const float* Wmix   = (const float*)d_in[6];

  const int BT = 2 * NB;                 // 4096 tokens

  char* ws = (char*)d_ws;
  size_t off = 0;
  auto carve = [&](size_t bytes) { char* p = ws + off; off = al256(off + bytes); return p; };

  bf16*  X     = (bf16*) carve((size_t)BT * DIMB * 2);           // tokens bf16
  bf16*  WqT   = (bf16*) carve((size_t)DIB * DIMB * 2);          // [512][1024]
  bf16*  WkvT  = (bf16*) carve((size_t)(2 * DIB) * DIMB * 2);    // [1024][1024]
  bf16*  WoT   = (bf16*) carve((size_t)DIMB * DIB * 2);          // [1024][512]
  float* gates = (float*)carve((size_t)BT * HEADSB * 4);
  float* mix   = (float*)carve((size_t)BT * HEADSB * 4);
  float* qraw  = (float*)carve((size_t)BT * DIB * 4);
  float* kvraw = (float*)carve((size_t)BT * 2 * DIB * 4);
  bf16*  qsb   = (bf16*) carve((size_t)2 * HEADSB * NB * DHB * 2);
  bf16*  ksb   = (bf16*) carve((size_t)2 * HEADSB * NB * DHB * 2);
  bf16*  vTb   = (bf16*) carve((size_t)2 * HEADSB * DHB * NB * 2);
  bf16*  outg  = (bf16*) carve((size_t)BT * DIB * 2);
  (void)ws_size; (void)in_sizes; (void)n_in; (void)out_size;

  // 1. conversions / weight transposes
  {
    int n = BT * DIMB;
    k_convert<<<(n + 255) / 256, 256, 0, stream>>>(tokens, X, n);
  }
  k_transpose<<<(DIMB * DIB + 255) / 256, 256, 0, stream>>>(Wq, WqT, DIMB, DIB);
  k_transpose<<<(DIMB * 2 * DIB + 255) / 256, 256, 0, stream>>>(Wkv, WkvT, DIMB, 2 * DIB);
  k_transpose<<<(DIB * DIMB + 255) / 256, 256, 0, stream>>>(Wo, WoT, DIB, DIMB);

  // 2. gates / mix
  k_gates<<<BT, 128, 0, stream>>>(tokens, Wg, Wmix, gates, mix);

  // 3. projections (BM=128 tiles)
  k_gemm<<<dim3(DIB / 64, BT / 128), 128, 0, stream>>>(X, WqT,  qraw,  BT, DIB,     DIMB);
  k_gemm<<<dim3(2 * DIB / 64, BT / 128), 128, 0, stream>>>(X, WkvT, kvraw, BT, 2 * DIB, DIMB);

  // 4. RoPE + value-residual epilogue
  {
    int n = 2 * HEADSB * NB * 32;
    k_qkv_epi<<<(n + 255) / 256, 256, 0, stream>>>(qraw, kvraw, vres, mix, qsb, ksb, vTb);
  }

  // 5. flash attention (window 512) with gate epilogue
  k_flash<<<dim3(NB / 64, 2 * HEADSB), 128, 0, stream>>>(qsb, ksb, vTb, gates, outg);

  // 6. output projection
  k_gemm<<<dim3(DIMB / 64, BT / 128), 128, 0, stream>>>(outg, WoT, (float*)d_out,
                                                        BT, DIMB, DIB);
}
